// Hyperedge_Attention_46471546143239
// MI455X (gfx1250) — matrix operand order repaired and checked by reading, not compiled, not run on hardware
//
#include <hip/hip_runtime.h>

typedef __attribute__((ext_vector_type(16))) _Float16 v16h;
typedef __attribute__((ext_vector_type(8)))  _Float16 v8h;
typedef __attribute__((ext_vector_type(8)))  float    v8f;

#define N_NODES 1048576
#define N_SEG   8192
#define D_IN    128
#define D_H     64

// Persistent-chunk shape for kernel A: 2048 blocks x 8 iters x 64 rows = 1M rows
#define BLOCKS_A 2048
#define ITERS_A  (N_NODES / (BLOCKS_A * 64))

// ---------------------------------------------------------------------------
// WMMA 16x16x32 f16 helper
// ---------------------------------------------------------------------------
__device__ __forceinline__ v8f wmma_f16(v16h a, v16h b, v8f c) {
  return __builtin_amdgcn_wmma_f32_16x16x32_f16(
      /*neg_a=*/false, a, /*neg_b=*/false, b,
      /*c_mod=*/(short)0, c, /*reuse_a=*/false, /*reuse_b=*/false);
}

// ---------------------------------------------------------------------------
// A-matrix (16x32, f16) loader per ISA 7.12.2:
//   lane l<16 : row m=l,  halves 0..7 -> K = kbase+0..7,  halves 8..15 -> K = kbase+16..23
//   lane l>=16: row m=l-16, halves 0..7 -> K = kbase+8..15, halves 8..15 -> K = kbase+24..31
// Source is row-major f32 with leading dim ld; converted to f16.
// ---------------------------------------------------------------------------
__device__ __forceinline__ v16h load_a_f32(const float* base, int ld, int kbase, int lane) {
  int m = lane & 15, g = lane >> 4;
  const float4* c0 = (const float4*)(base + m * ld + kbase + 8 * g);
  const float4* c1 = (const float4*)(base + m * ld + kbase + 16 + 8 * g);
  float4 t0 = c0[0], t1 = c0[1];
  float4 t2 = c1[0], t3 = c1[1];
  v16h a;
  a[0] = (_Float16)t0.x; a[1] = (_Float16)t0.y; a[2] = (_Float16)t0.z; a[3] = (_Float16)t0.w;
  a[4] = (_Float16)t1.x; a[5] = (_Float16)t1.y; a[6] = (_Float16)t1.z; a[7] = (_Float16)t1.w;
  a[8]  = (_Float16)t2.x; a[9]  = (_Float16)t2.y; a[10] = (_Float16)t2.z; a[11] = (_Float16)t2.w;
  a[12] = (_Float16)t3.x; a[13] = (_Float16)t3.y; a[14] = (_Float16)t3.z; a[15] = (_Float16)t3.w;
  return a;
}

// Same A layout, but reading already-f16 data (from LDS).
__device__ __forceinline__ v16h load_a_f16(const _Float16* base, int ld, int kbase, int lane) {
  int m = lane & 15, g = lane >> 4;
  v8h t0 = *(const v8h*)(base + m * ld + kbase + 8 * g);
  v8h t1 = *(const v8h*)(base + m * ld + kbase + 16 + 8 * g);
  v16h a;
#pragma unroll
  for (int i = 0; i < 8; ++i) { a[i] = t0[i]; a[8 + i] = t1[i]; }
  return a;
}

// ---------------------------------------------------------------------------
// B-matrix (32x16, f16) loader. B[k][n] = W[n][k] (W row-major NxK, f16, LDS).
//   lane l<16 : column n=l,  halves 0..7 -> K = kbase+0..7,  halves 8..15 -> K = kbase+16..23
//   lane l>=16: column n=l-16, halves 0..7 -> K = kbase+8..15, halves 8..15 -> K = kbase+24..31
// ---------------------------------------------------------------------------
__device__ __forceinline__ v16h load_b_f16(const _Float16* W, int ldw, int ntile, int kbase, int lane) {
  int n = lane & 15, g = lane >> 4;
  v8h t0 = *(const v8h*)(W + (ntile * 16 + n) * ldw + kbase + 8 * g);
  v8h t1 = *(const v8h*)(W + (ntile * 16 + n) * ldw + kbase + 16 + 8 * g);
  v16h b;
#pragma unroll
  for (int i = 0; i < 8; ++i) { b[i] = t0[i]; b[8 + i] = t1[i]; }
  return b;
}

// searchsorted(offset, i, side='right') - 1 : last j with offset[j] <= i
__device__ __forceinline__ int seg_of(const int* __restrict__ offset, int i) {
  int lo = 0, hi = N_SEG;
#pragma unroll 1
  while (lo < hi) {
    int mid = (lo + hi + 1) >> 1;
    if (offset[mid] <= i) lo = mid; else hi = mid - 1;
  }
  return lo;
}

// ---------------------------------------------------------------------------
// Kernel 0: zero the tokens accumulator (d_ws is not re-poisoned per replay)
// ---------------------------------------------------------------------------
__global__ void zero_tokens_kernel(float* __restrict__ p, int n) {
  int i = blockIdx.x * blockDim.x + threadIdx.x;
  if (i < n) p[i] = 0.0f;
}

// ---------------------------------------------------------------------------
// Kernel A: fused MLP (x @ W1^T + b1 -> relu -> @ W2^T + b2 -> relu)
//           + run-compressed segment-sum via global_atomic_add_f32.
// 128 threads = 4 waves; each wave computes 16-row x 64-col tiles.
// Persistent over ITERS_A row-tiles so weights are staged in LDS only once
// per 512 rows; next tile's x is prefetched (global_prefetch_b8) each iter.
// ---------------------------------------------------------------------------
__global__ void __launch_bounds__(128)
mlp_segsum_kernel(const float* __restrict__ x, const int* __restrict__ offset,
                  const float* __restrict__ W1, const float* __restrict__ b1,
                  const float* __restrict__ W2, const float* __restrict__ b2,
                  float* __restrict__ tokens) {
  __shared__ __align__(32) _Float16 W1s[D_H * D_IN];   // 16 KB
  __shared__ __align__(32) _Float16 W2s[D_H * D_H];    //  8 KB
  __shared__ __align__(32) _Float16 h1s[4][16 * D_H];  //  8 KB (per-wave transpose buffer)
  __shared__ int segs[4][16];

  const int tid  = threadIdx.x;
  const int wid  = tid >> 5;
  const int lane = tid & 31;
  const int g    = lane >> 4;
  const int nc   = lane & 15;

  // Stage weights as f16 into LDS (cooperative, once per block).
#pragma unroll 4
  for (int i = tid; i < D_H * D_IN; i += 128) W1s[i] = (_Float16)W1[i];
#pragma unroll 2
  for (int i = tid; i < D_H * D_H; i += 128) W2s[i] = (_Float16)W2[i];
  __syncthreads();

  const int chunkBase = blockIdx.x * (ITERS_A * 64);

#pragma unroll 1
  for (int it = 0; it < ITERS_A; ++it) {
    const int rowBase = chunkBase + it * 64 + wid * 16;

    // Prefetch next iteration's 64-row x-tile (32 KB): 2 cachelines/thread.
    if (it + 1 < ITERS_A) {
      const float* xn = x + (long long)(chunkBase + (it + 1) * 64) * D_IN;
      __builtin_prefetch(xn + tid * 64, 0, 1);
      __builtin_prefetch(xn + tid * 64 + 32, 0, 1);
    }

    // Per-row segment ids (rows are contiguous; offsets sorted).
    if (lane < 16) segs[wid][lane] = seg_of(offset, rowBase + lane);

    // ---- GEMM1: h1 = relu(x(16x128) @ W1^T + b1) -> 16x64 ----
    v16h aX[4];
#pragma unroll
    for (int ks = 0; ks < 4; ++ks)
      aX[ks] = load_a_f32(x + (long long)rowBase * D_IN, D_IN, ks * 32, lane);

#pragma unroll
    for (int n = 0; n < 4; ++n) {
      v8f acc = {};
#pragma unroll
      for (int ks = 0; ks < 4; ++ks) {
        v16h b = load_b_f16(W1s, D_IN, n, ks * 32, lane);
        acc = wmma_f16(aX[ks], b, acc);
      }
      float bias = b1[n * 16 + nc];
#pragma unroll
      for (int r = 0; r < 8; ++r) {
        float v = fmaxf(acc[r] + bias, 0.0f);
        h1s[wid][(r + 8 * g) * D_H + n * 16 + nc] = (_Float16)v;  // row-major 16x64
      }
    }

    // ---- GEMM2: h2 = relu(h1(16x64) @ W2^T + b2) ----
    v16h aH[2];
#pragma unroll
    for (int ks = 0; ks < 2; ++ks)
      aH[ks] = load_a_f16(&h1s[wid][0], D_H, ks * 32, lane);

    const int sFirst = segs[wid][8 * g];
    const int sLast  = segs[wid][8 * g + 7];

#pragma unroll
    for (int n = 0; n < 4; ++n) {
      v8f acc = {};
#pragma unroll
      for (int ks = 0; ks < 2; ++ks) {
        v16h b = load_b_f16(W2s, D_H, n, ks * 32, lane);
        acc = wmma_f16(aH[ks], b, acc);
      }
      float bias = b2[n * 16 + nc];
      const int col = n * 16 + nc;
      if (sFirst == sLast) {
        // all 8 rows owned by this lane land in one segment: one atomic
        float s = 0.0f;
#pragma unroll
        for (int r = 0; r < 8; ++r) s += fmaxf(acc[r] + bias, 0.0f);
        atomicAdd(&tokens[sFirst * D_H + col], s);
      } else {
#pragma unroll
        for (int r = 0; r < 8; ++r) {
          int s = segs[wid][r + 8 * g];
          atomicAdd(&tokens[s * D_H + col], fmaxf(acc[r] + bias, 0.0f));
        }
      }
    }
  }
}

// ---------------------------------------------------------------------------
// Kernel B: softmax over a length-1 axis is identity, so
//   out = (tokens @ Wv^T + bv) @ out_w^T + out_b
// ---------------------------------------------------------------------------
__global__ void __launch_bounds__(128)
attn_out_kernel(const float* __restrict__ tokens,
                const float* __restrict__ in_proj_w, const float* __restrict__ in_proj_b,
                const float* __restrict__ out_w, const float* __restrict__ out_b,
                float* __restrict__ out) {
  __shared__ __align__(32) _Float16 Wvs[D_H * D_H];   // 8 KB
  __shared__ __align__(32) _Float16 Wos[D_H * D_H];   // 8 KB
  __shared__ __align__(32) _Float16 vts[4][16 * D_H]; // 8 KB

  const int tid  = threadIdx.x;
  const int wid  = tid >> 5;
  const int lane = tid & 31;

  const float* Wv = in_proj_w + 2 * D_H * D_H;  // rows [128,192) = Wv
  const float* bv = in_proj_b + 2 * D_H;

#pragma unroll 2
  for (int i = tid; i < D_H * D_H; i += 128) {
    Wvs[i] = (_Float16)Wv[i];
    Wos[i] = (_Float16)out_w[i];
  }
  __syncthreads();

  const int rowBase = (blockIdx.x * 4 + wid) * 16;
  const int g  = lane >> 4;
  const int nc = lane & 15;

  // v = tokens(16x64) @ Wv^T + bv
  v16h aT[2];
#pragma unroll
  for (int ks = 0; ks < 2; ++ks)
    aT[ks] = load_a_f32(tokens + (long long)rowBase * D_H, D_H, ks * 32, lane);

#pragma unroll
  for (int n = 0; n < 4; ++n) {
    v8f acc = {};
#pragma unroll
    for (int ks = 0; ks < 2; ++ks) {
      v16h b = load_b_f16(Wvs, D_H, n, ks * 32, lane);
      acc = wmma_f16(aT[ks], b, acc);
    }
    float bias = bv[n * 16 + nc];
#pragma unroll
    for (int r = 0; r < 8; ++r)
      vts[wid][(r + 8 * g) * D_H + n * 16 + nc] = (_Float16)(acc[r] + bias);
  }

  // out = v(16x64) @ out_w^T + out_b
  v16h aV[2];
#pragma unroll
  for (int ks = 0; ks < 2; ++ks)
    aV[ks] = load_a_f16(&vts[wid][0], D_H, ks * 32, lane);

#pragma unroll
  for (int n = 0; n < 4; ++n) {
    v8f acc = {};
#pragma unroll
    for (int ks = 0; ks < 2; ++ks) {
      v16h b = load_b_f16(Wos, D_H, n, ks * 32, lane);
      acc = wmma_f16(aV[ks], b, acc);
    }
    float bias = out_b[n * 16 + nc];
#pragma unroll
    for (int r = 0; r < 8; ++r)
      out[(long long)(rowBase + r + 8 * g) * D_H + n * 16 + nc] = acc[r] + bias;
  }
}

// ---------------------------------------------------------------------------
extern "C" void kernel_launch(void* const* d_in, const int* in_sizes, int n_in,
                              void* d_out, int out_size, void* d_ws, size_t ws_size,
                              hipStream_t stream) {
  const float* x         = (const float*)d_in[0];
  const int*   offset    = (const int*)d_in[1];
  const float* W1        = (const float*)d_in[2];
  const float* b1        = (const float*)d_in[3];
  const float* W2        = (const float*)d_in[4];
  const float* b2        = (const float*)d_in[5];
  const float* in_proj_w = (const float*)d_in[6];
  const float* in_proj_b = (const float*)d_in[7];
  const float* out_w     = (const float*)d_in[8];
  const float* out_b     = (const float*)d_in[9];
  float* out    = (float*)d_out;
  float* tokens = (float*)d_ws;  // N_SEG * D_H f32 accumulator (2 MB)

  zero_tokens_kernel<<<(N_SEG * D_H + 255) / 256, 256, 0, stream>>>(tokens, N_SEG * D_H);

  mlp_segsum_kernel<<<BLOCKS_A, 128, 0, stream>>>(x, offset, W1, b1, W2, b2, tokens);

  attn_out_kernel<<<N_SEG / 64, 128, 0, stream>>>(tokens, in_proj_w, in_proj_b,
                                                  out_w, out_b, out);
}
